// How2comm_32427003085275
// MI455X (gfx1250) — matrix operand order repaired and checked by reading, not compiled, not run on hardware
//
#include <hip/hip_runtime.h>

typedef __attribute__((ext_vector_type(16))) _Float16 v16h;
typedef __attribute__((ext_vector_type(8)))  float    v8f;

constexpr int Bb = 4, Ll = 5, Cc = 64, Hh = 100, Ww = 352;
constexpr int HWW = Hh * Ww;
constexpr int KSTEPS = 18;                          // Cin*9 / 32 = 576/32
constexpr int WPACK_HALVES = KSTEPS * 64 * 2 * 16;  // 36864 per conv
constexpr int PTILE = 64;                           // pixels per block (4 N-tiles)
constexpr int PCOLS = PTILE + 2;                    // patch row width (66)

// ---------------------------------------------------------------- theta ----
__global__ void theta_kernel(const float* __restrict__ pt, float* __restrict__ theta) {
  int t = threadIdx.x;
  if (t >= Bb * Ll) return;
  int b = t / Ll, l = t % Ll;
  const float* s = pt + (((size_t)b * Ll + 0) * Ll + l) * 16;  // t[b,0,l] 4x4
  float* o = theta + t * 6;
  o[0] = s[0];
  o[1] = s[1] * ((float)Hh / (float)Ww);
  o[2] = s[3] * (2.0f / (2.0f * 0.4f * (float)Ww));
  o[3] = s[4] * ((float)Ww / (float)Hh);
  o[4] = s[5];
  o[5] = s[7] * (2.0f / (2.0f * 0.4f * (float)Hh));
}

// ------------------------------------------------- weight repack (A frag) ----
// wpack[conv][ks][cout][hi][j] = w[cout][cin][ky][kx]*g[cout]
// with k = ks*32 + (j<8 ? j : j+8) + 8*hi, tap = k>>6, cin = k&63
__global__ void repack_kernel(const float* __restrict__ vw, const float* __restrict__ vg,
                              const float* __restrict__ fw, const float* __restrict__ fg,
                              _Float16* __restrict__ wpack) {
  int gid = blockIdx.x * 256 + threadIdx.x;
  if (gid >= 2 * WPACK_HALVES) return;
  int conv = gid / WPACK_HALVES;
  int e    = gid % WPACK_HALVES;
  int j    = e & 15;
  int hi   = (e >> 4) & 1;
  int cout = (e >> 5) & 63;
  int ks   = e >> 11;
  int k    = ks * 32 + ((j < 8) ? j : j + 8) + 8 * hi;
  int tap  = k >> 6;
  int cin  = k & 63;
  int ky = tap / 3, kx = tap % 3;
  const float* w = conv ? fw : vw;
  const float* g = conv ? fg : vg;
  wpack[gid] = (_Float16)(w[((cout * 64 + cin) * 3 + ky) * 3 + kx] * g[cout]);
}

// ------------------------------------------- warp + max over L, f16 store ----
__global__ void warp_max_kernel(const float* __restrict__ x,
                                const float* __restrict__ vox,
                                const float* __restrict__ theta,
                                _Float16* __restrict__ aggF,
                                _Float16* __restrict__ aggV) {
  int w0 = blockIdx.x * 32, h = blockIdx.y, b = blockIdx.z;
  int t = threadIdx.x;
  int pix = t & 31, cg = t >> 5;
  int w = w0 + pix;
  float gx = -1.0f + 2.0f * (float)w / (float)(Ww - 1);
  float gy = -1.0f + 2.0f * (float)h / (float)(Hh - 1);
  float accF[8], accV[8];
#pragma unroll
  for (int i = 0; i < 8; ++i) { accF[i] = -INFINITY; accV[i] = -INFINITY; }
  for (int l = 0; l < Ll; ++l) {
    const float* th = theta + (b * Ll + l) * 6;
    float sx = th[0] * gx + th[1] * gy + th[2];
    float sy = th[3] * gx + th[4] * gy + th[5];
    float px = (sx + 1.0f) * 0.5f * (float)(Ww - 1);
    float py = (sy + 1.0f) * 0.5f * (float)(Hh - 1);
    float x0 = floorf(px), y0 = floorf(py);
    float x1 = x0 + 1.0f, y1 = y0 + 1.0f;
    float wx1 = px - x0, wx0 = 1.0f - wx1;
    float wy1 = py - y0, wy0 = 1.0f - wy1;
    float vx0 = (x0 >= 0.f && x0 <= (float)(Ww - 1)) ? 1.f : 0.f;
    float vx1 = (x1 >= 0.f && x1 <= (float)(Ww - 1)) ? 1.f : 0.f;
    float vy0 = (y0 >= 0.f && y0 <= (float)(Hh - 1)) ? 1.f : 0.f;
    float vy1 = (y1 >= 0.f && y1 <= (float)(Hh - 1)) ? 1.f : 0.f;
    int xc0 = (int)fminf(fmaxf(x0, 0.f), (float)(Ww - 1));
    int xc1 = (int)fminf(fmaxf(x1, 0.f), (float)(Ww - 1));
    int yc0 = (int)fminf(fmaxf(y0, 0.f), (float)(Hh - 1));
    int yc1 = (int)fminf(fmaxf(y1, 0.f), (float)(Hh - 1));
    float w00 = wx0 * wy0 * vx0 * vy0, w10 = wx1 * wy0 * vx1 * vy0;
    float w01 = wx0 * wy1 * vx0 * vy1, w11 = wx1 * wy1 * vx1 * vy1;
    int o00 = yc0 * Ww + xc0, o10 = yc0 * Ww + xc1;
    int o01 = yc1 * Ww + xc0, o11 = yc1 * Ww + xc1;
    int nb = b * Ll + l;
#pragma unroll
    for (int c8 = 0; c8 < 8; ++c8) {
      int c = cg * 8 + c8;
      const float* fp = x   + (size_t)(nb * Cc + c) * HWW;
      const float* vp = vox + (size_t)(nb * Cc + c) * HWW;
      float vF = fp[o00] * w00 + fp[o10] * w10 + fp[o01] * w01 + fp[o11] * w11;
      float vV = vp[o00] * w00 + vp[o10] * w10 + vp[o01] * w01 + vp[o11] * w11;
      accF[c8] = fmaxf(accF[c8], vF);
      accV[c8] = fmaxf(accV[c8], vV);
    }
  }
#pragma unroll
  for (int c8 = 0; c8 < 8; ++c8) {
    int c = cg * 8 + c8;
    size_t oi = ((size_t)(b * Cc + c) * Hh + h) * Ww + w;
    aggF[oi] = (_Float16)accF[c8];
    aggV[oi] = (_Float16)accV[c8];
  }
}

// ------------------------------------------- implicit-GEMM 3x3 conv (WMMA) ----
// Each wave: one N-tile (16 px), TWO M-tiles (32 Cout) sharing each B fragment.
__device__ __forceinline__ void conv_gemm(const _Float16* __restrict__ patch,
                                          const _Float16* __restrict__ wpk,
                                          const float* __restrict__ biasS,
                                          float* __restrict__ outBuf,
                                          int waveId, int lane) {
  int nt  = waveId & 3;     // pixel tile (4 x 16)
  int mtg = waveId >> 2;    // Cout half (2 x 32)
  int lm = lane & 15;
  int hi = lane >> 4;
  int cbase = mtg * 32;
  v8f acc0 = {}, acc1 = {};
#pragma unroll
  for (int ks = 0; ks < KSTEPS; ++ks) {
    // B fragment: lane = N (pixel), element j = K in table order (same as A)
    int n = nt * 16 + lm;
    v16h bf;
#pragma unroll
    for (int j = 0; j < 16; ++j) {
      int k   = ks * 32 + ((j < 8) ? j : j + 8) + 8 * hi;
      int tap = k >> 6;
      int cin = k & 63;
      int ky = tap / 3, kx = tap % 3;
      bf[j] = patch[(cin * 3 + ky) * PCOLS + n + kx];
    }
    // A fragments: aligned 32B contiguous per lane from pre-permuted weights
    v16h a0 = *(const v16h*)(wpk + (((ks * 64 + cbase +      lm) * 2 + hi) << 4));
    v16h a1 = *(const v16h*)(wpk + (((ks * 64 + cbase + 16 + lm) * 2 + hi) << 4));
    acc0 = __builtin_amdgcn_wmma_f32_16x16x32_f16(false, a0, false, bf,
                                                  (short)0, acc0, false, false);
    acc1 = __builtin_amdgcn_wmma_f32_16x16x32_f16(false, a1, false, bf,
                                                  (short)0, acc1, false, false);
  }
#pragma unroll
  for (int r = 0; r < 8; ++r) {
    int m0 = cbase + r + 8 * hi;        // per ISA C/D layout
    int m1 = m0 + 16;
    int n  = nt * 16 + (lane & 15);
    float v0 = acc0[r] + biasS[m0];
    float v1 = acc1[r] + biasS[m1];
    outBuf[m0 * PTILE + n] = v0 > 0.f ? v0 : 0.f;
    outBuf[m1 * PTILE + n] = v1 > 0.f ? v1 : 0.f;
  }
}

// --------------------------------- conv x2 + attention + softmax + fuse ----
__global__ void conv_fuse_kernel(const float* __restrict__ x,
                                 const _Float16* __restrict__ aggV,
                                 const _Float16* __restrict__ aggF,
                                 const _Float16* __restrict__ wpack,
                                 const float* __restrict__ vb, const float* __restrict__ vg,
                                 const float* __restrict__ vbt,
                                 const float* __restrict__ fb, const float* __restrict__ fg,
                                 const float* __restrict__ fbt,
                                 const float* __restrict__ aw1, const float* __restrict__ ab1,
                                 const float* __restrict__ aw2, const float* __restrict__ ab2,
                                 float* __restrict__ out) {
  __shared__ _Float16 patch[64 * 3 * PCOLS];   // 25344 B
  __shared__ float veBuf[64 * PTILE];          // 16 KB
  __shared__ float feBuf[64 * PTILE];          // 16 KB
  __shared__ float aw1s[32 * 128];             // 16 KB
  __shared__ float a1Buf[32 * PTILE];          // 8 KB
  __shared__ float biasV[64], biasF[64];
  __shared__ float wBuf[PTILE];

  int w0 = blockIdx.x * PTILE, h = blockIdx.y, b = blockIdx.z;
  int t = threadIdx.x;
  int lane = t & 31, waveId = t >> 5;

  for (int e = t; e < 4096; e += 256) aw1s[e] = aw1[e];
  if (t < 64) {
    biasV[t] = vb[t] * vg[t] + vbt[t];
    biasF[t] = fb[t] * fg[t] + fbt[t];
  }
  // stage vox patch (zero pad at image borders and past W)
  for (int e = t; e < 64 * 3 * PCOLS; e += 256) {
    int c = e / (3 * PCOLS), rem = e % (3 * PCOLS), ry = rem / PCOLS, col = rem % PCOLS;
    int hin = h + ry - 1, win = w0 + col - 1;
    _Float16 v = (_Float16)0.f;
    if (hin >= 0 && hin < Hh && win >= 0 && win < Ww)
      v = aggV[((size_t)(b * Cc + c) * Hh + hin) * Ww + win];
    patch[e] = v;
  }
  __syncthreads();
  conv_gemm(patch, wpack, biasV, veBuf, waveId, lane);
  __syncthreads();
  // stage feat patch
  for (int e = t; e < 64 * 3 * PCOLS; e += 256) {
    int c = e / (3 * PCOLS), rem = e % (3 * PCOLS), ry = rem / PCOLS, col = rem % PCOLS;
    int hin = h + ry - 1, win = w0 + col - 1;
    _Float16 v = (_Float16)0.f;
    if (hin >= 0 && hin < Hh && win >= 0 && win < Ww)
      v = aggF[((size_t)(b * Cc + c) * Hh + hin) * Ww + win];
    patch[e] = v;
  }
  __syncthreads();
  conv_gemm(patch, wpack + WPACK_HALVES, biasF, feBuf, waveId, lane);
  __syncthreads();

  // attention hidden layer: 32 channels, 128-wide dot per pixel
  {
    int p = t & (PTILE - 1), part = t >> 6;   // 4 parts x 8 channels
#pragma unroll
    for (int cc = 0; cc < 8; ++cc) {
      int c1 = part * 8 + cc;
      float s = ab1[c1];
      const float* wrow = &aw1s[c1 * 128];
      for (int k = 0; k < 64; ++k) s += wrow[k]      * veBuf[k * PTILE + p];
      for (int k = 0; k < 64; ++k) s += wrow[64 + k] * feBuf[k * PTILE + p];
      a1Buf[c1 * PTILE + p] = s > 0.f ? s : 0.f;
    }
  }
  __syncthreads();
  if (t < PTILE) {
    float z0 = ab2[0], z1 = ab2[1];
    for (int k = 0; k < 32; ++k) {
      float av = a1Buf[k * PTILE + t];
      z0 += aw2[k] * av;
      z1 += aw2[32 + k] * av;
    }
    float m = fmaxf(z0, z1);
    float e0 = __expf(z0 - m), e1 = __expf(z1 - m);
    wBuf[t] = e0 / (e0 + e1);
  }
  __syncthreads();
#pragma unroll
  for (int i = 0; i < 16; ++i) {
    int q = t + i * 256;
    int c = q >> 6, p = q & (PTILE - 1);
    int w = w0 + p;
    if (w < Ww) {
      float a0 = wBuf[p];
      float fused = a0 * veBuf[c * PTILE + p] + (1.f - a0) * feBuf[c * PTILE + p];
      float ego = x[(size_t)((b * Ll + 0) * Cc + c) * HWW + h * Ww + w];
      out[((size_t)(b * Cc + c) * Hh + h) * Ww + w] = ego + fused;
    }
  }
}

// ------------------------------------------------------------------ launch ----
extern "C" void kernel_launch(void* const* d_in, const int* in_sizes, int n_in,
                              void* d_out, int out_size, void* d_ws, size_t ws_size,
                              hipStream_t stream) {
  const float* x   = (const float*)d_in[0];
  const float* vox = (const float*)d_in[1];
  // d_in[2] = det_bev: multiplied by 0.0 in reference -> skipped
  const float* pt  = (const float*)d_in[3];
  // d_in[4] = record_len: unused
  const float* vw  = (const float*)d_in[5];
  const float* vb  = (const float*)d_in[6];
  const float* vg  = (const float*)d_in[7];
  const float* vbt = (const float*)d_in[8];
  const float* fw  = (const float*)d_in[9];
  const float* fb  = (const float*)d_in[10];
  const float* fg  = (const float*)d_in[11];
  const float* fbt = (const float*)d_in[12];
  const float* aw1 = (const float*)d_in[13];
  const float* ab1 = (const float*)d_in[14];
  const float* aw2 = (const float*)d_in[15];
  const float* ab2 = (const float*)d_in[16];
  float* out = (float*)d_out;

  // workspace layout
  float*    theta = (float*)d_ws;                              // 480 B
  _Float16* wpack = (_Float16*)((char*)d_ws + 512);            // 2*72 KB
  _Float16* aggV  = (_Float16*)((char*)d_ws + 512 + 2 * WPACK_HALVES * 2);
  _Float16* aggF  = aggV + (size_t)Bb * Cc * Hh * Ww;          // 18 MB each

  theta_kernel<<<1, 32, 0, stream>>>(pt, theta);
  repack_kernel<<<(2 * WPACK_HALVES + 255) / 256, 256, 0, stream>>>(vw, vg, fw, fg, wpack);
  dim3 gridW(Ww / 32, Hh, Bb);                 // 11 x 100 x 4
  warp_max_kernel<<<gridW, 256, 0, stream>>>(x, vox, theta, aggF, aggV);
  dim3 gridC((Ww + PTILE - 1) / PTILE, Hh, Bb);  // 6 x 100 x 4 (last tile masked)
  conv_fuse_kernel<<<gridC, 256, 0, stream>>>(x, aggV, aggF, wpack,
                                              vb, vg, vbt, fb, fg, fbt,
                                              aw1, ab1, aw2, ab2, out);
}
